// LSTMGraphSAGE_3504693313810
// MI455X (gfx1250) — matrix operand order repaired and checked by reading, not compiled, not run on hardware
//
#include <hip/hip_runtime.h>
#include <cstdint>
#include <cstddef>

// ---------------------------------------------------------------------------
// CDNA5 (gfx1250) LSTM-GraphSAGE. wave32, WMMA bf16 16x16x32 w/ f32 accum.
// LDS activations kept as packed bf16 pairs so WMMA A-fragment build is pure
// ds_load_b32 (the 16-bit A layout's dword pairs are K-adjacent).
// WMMA issued as independent accumulator chains (split ih/hh accumulators +
// kt-outer/tile-inner interleave) to avoid 7.12.1 WMMA->WMMA RAW hazard NOPs.
// ---------------------------------------------------------------------------

typedef __attribute__((ext_vector_type(16))) __bf16    v16bf;
typedef __attribute__((ext_vector_type(8)))  float     v8f;
typedef __attribute__((ext_vector_type(4)))  unsigned  u32x4;

#define NNODES 32768

__device__ __forceinline__ unsigned short f2bf(float f) {
  unsigned x = __float_as_uint(f);
  return (unsigned short)((x + 0x7FFFu + ((x >> 16) & 1u)) >> 16);
}
__device__ __forceinline__ unsigned pack2bf(float f0, float f1) {
  return (unsigned)f2bf(f0) | ((unsigned)f2bf(f1) << 16);
}
__device__ __forceinline__ float bf2f(unsigned short b) {
  return __uint_as_float(((unsigned)b) << 16);
}
// fast activations: v_exp_f32 + v_rcp_f32 (no precise-division NR chains)
__device__ __forceinline__ float sigf(float x) {
  return __builtin_amdgcn_rcpf(1.f + __expf(-x));
}
__device__ __forceinline__ float tanh_f(float x) {
  x = fminf(fmaxf(x, -15.f), 15.f);
  float e = __expf(2.f * x);
  return (e - 1.f) * __builtin_amdgcn_rcpf(e + 1.f);
}

// A-matrix fragment (16x32 bf16) from a row-major f32 row (global memory).
// ISA layout (16-bit A 16x32): lane half h: elems 0..7 -> K = 8h+e,
// elems 8..15 -> K = 16 + 8h + (e-8).
template <int KACT>
__device__ __forceinline__ v16bf a_frag_f32(const float* __restrict__ row, int kt, int half) {
  union { v16bf v; unsigned u[8]; } u;
#pragma unroll
  for (int d = 0; d < 8; ++d) {
    int e = 2 * d;
    int k0 = kt * 32 + ((e < 8) ? (8 * half + e) : (16 + 8 * half + (e - 8)));
    float f0 = (k0 < KACT) ? row[k0] : 0.f;
    float f1 = (k0 + 1 < KACT) ? row[k0 + 1] : 0.f;
    u.u[d] = pack2bf(f0, f1);
  }
  return u.v;
}

// A-matrix fragment from packed-bf16 LDS row (1 dword = 2 adjacent K elems).
__device__ __forceinline__ v16bf a_frag_pk(const unsigned* rowpk, int kt, int half) {
  union { v16bf v; unsigned u[8]; } u;
#pragma unroll
  for (int d = 0; d < 8; ++d) {
    int e = 2 * d;
    int k0 = kt * 32 + ((e < 8) ? (8 * half + e) : (16 + 8 * half + (e - 8)));
    u.u[d] = rowpk[k0 >> 1];
  }
  return u.v;
}

// B-matrix fragment: pre-packed in global memory, 8 dwords per lane.
__device__ __forceinline__ v16bf b_frag(const unsigned* __restrict__ fb,
                                        int gt, int kt, int KT, int lane) {
  const unsigned* p = fb + ((size_t)(gt * KT + kt) * 32 + lane) * 8;
  union { v16bf v; u32x4 q[2]; } u;
  u.q[0] = *(const u32x4*)p;
  u.q[1] = *(const u32x4*)(p + 4);
  return u.v;
}

__device__ __forceinline__ v8f wmma_bf16(v16bf a, v16bf b, v8f c) {
  return __builtin_amdgcn_wmma_f32_16x16x32_bf16(false, a, false, b, (short)0, c,
                                                 false, false);
}

// ---------------------------------------------------------------------------
// Pack W (GxK row-major f32) into WMMA B-fragment layout (bf16, K padded to
// KT*32). B layout matches A: lane half h: elems 0..7 -> K = 8h+e,
// elems 8..15 -> K = 16 + 8h + (e-8); lanes 0-15 / 16-31 = same cols.
// ---------------------------------------------------------------------------
__global__ __launch_bounds__(32) void convert_wfrag_kernel(
    const float* __restrict__ W, unsigned* __restrict__ dst, int G, int K, int KT) {
  int lane = threadIdx.x & 31;
  int gt = blockIdx.x, kt = blockIdx.y;
  int g = gt * 16 + (lane & 15);
  int half = lane >> 4;
  unsigned out[8];
#pragma unroll
  for (int d = 0; d < 8; ++d) {
    int k0 = kt * 32 + ((d < 4) ? (8 * half + 2 * d) : (16 + 8 * half + 2 * (d - 4)));
    int k1 = k0 + 1;
    float f0 = (k0 < K) ? W[(size_t)g * K + k0] : 0.f;
    float f1 = (k1 < K) ? W[(size_t)g * K + k1] : 0.f;
    out[d] = pack2bf(f0, f1);
  }
  unsigned* p = dst + ((size_t)(gt * KT + kt) * 32 + lane) * 8;
  u32x4 lo = {out[0], out[1], out[2], out[3]};
  u32x4 hi = {out[4], out[5], out[6], out[7]};
  *(u32x4*)p = lo;
  *(u32x4*)(p + 4) = hi;
}

// ---------------------------------------------------------------------------
// Fused neighbor-LSTM (LSTM_r): 16-node tile per block, 16 time steps,
// state in LDS (packed bf16 for WMMA operands, f32 cell state),
// gates via WMMA. Writes final h (N x FI).
// ---------------------------------------------------------------------------
template <int FI, int G1, int KT, int NW>
__global__ __launch_bounds__(NW * 32) void lstm_r_kernel(
    const float* __restrict__ xin, const int* __restrict__ nbr,
    const unsigned* __restrict__ fih, const unsigned* __restrict__ fhh,
    const float* __restrict__ bih, const float* __restrict__ bhh,
    float* __restrict__ hT_out) {
  constexpr int FIP = KT * 32;
  constexpr int TPW = G1 / (16 * NW);  // gate tiles per wave
  __shared__ unsigned sm_pk[16][FIP / 2];  // gathered rows, packed bf16
  __shared__ unsigned sh_pk[16][FIP / 2];  // hidden state, packed bf16
  __shared__ float    sc[16][FI];          // cell state (f32)
  __shared__ float    sg[16][G1];          // gates
  __shared__ float    sbias[G1];

  const int tid = threadIdx.x;
  const int lane = tid & 31;
  const int wave = tid >> 5;
  const int node0 = blockIdx.x * 16;
  const int r = lane & 15, half = lane >> 4;

  for (int i = tid; i < 16 * (FIP / 2); i += NW * 32) {
    (&sm_pk[0][0])[i] = 0u;
    (&sh_pk[0][0])[i] = 0u;
  }
  for (int i = tid; i < 16 * FI; i += NW * 32) (&sc[0][0])[i] = 0.f;
  for (int g = tid; g < G1; g += NW * 32) sbias[g] = bih[g] + bhh[g];
  __syncthreads();

  for (int t = 0; t < 16; ++t) {
    // gather m_t rows into LDS (convert once, packed bf16)
    for (int i = tid; i < 16 * (FI / 2); i += NW * 32) {
      int rr = i / (FI / 2), kk = i % (FI / 2);
      int src = nbr[(size_t)(node0 + rr) * 16 + t];
      float f0 = xin[(size_t)src * FI + 2 * kk];
      float f1 = xin[(size_t)src * FI + 2 * kk + 1];
      sm_pk[rr][kk] = pack2bf(f0, f1);
    }
    __syncthreads();

    // build A fragments (input + hidden): pure ds_load_b32
    v16bf a1[KT], a2[KT];
#pragma unroll
    for (int kt = 0; kt < KT; ++kt) {
      a1[kt] = a_frag_pk(&sm_pk[r][0], kt, half);
      a2[kt] = a_frag_pk(&sh_pk[r][0], kt, half);
    }

    // gates = m_t @ Wih^T + h @ Whh^T
    // Separate ih/hh accumulator sets + kt-outer/tile-inner interleave:
    // every adjacent WMMA pair is independent -> no RAW hazard NOPs.
    v8f accA[TPW] = {};
    v8f accB[TPW] = {};
#pragma unroll
    for (int kt = 0; kt < KT; ++kt) {
#pragma unroll
      for (int j = 0; j < TPW; ++j)
        accA[j] = wmma_bf16(a1[kt], b_frag(fih, wave * TPW + j, kt, KT, lane), accA[j]);
#pragma unroll
      for (int j = 0; j < TPW; ++j)
        accB[j] = wmma_bf16(a2[kt], b_frag(fhh, wave * TPW + j, kt, KT, lane), accB[j]);
    }
#pragma unroll
    for (int j = 0; j < TPW; ++j) {
      int col = (wave * TPW + j) * 16 + r;
      int mbase = half * 8;
      float bb = sbias[col];
#pragma unroll
      for (int m = 0; m < 8; ++m)
        sg[mbase + m][col] = accA[j][m] + accB[j][m] + bb;
    }
    __syncthreads();

    // LSTM cell update (i, f, g, o splits), two adjacent units per thread
    for (int i = tid; i < 16 * (FI / 2); i += NW * 32) {
      int rr = i / (FI / 2), kk = i % (FI / 2);
      int u0 = 2 * kk, u1 = u0 + 1;
      float cv0 = sigf(sg[rr][FI + u0]) * sc[rr][u0] +
                  sigf(sg[rr][u0]) * tanh_f(sg[rr][2 * FI + u0]);
      float hv0 = sigf(sg[rr][3 * FI + u0]) * tanh_f(cv0);
      float cv1 = sigf(sg[rr][FI + u1]) * sc[rr][u1] +
                  sigf(sg[rr][u1]) * tanh_f(sg[rr][2 * FI + u1]);
      float hv1 = sigf(sg[rr][3 * FI + u1]) * tanh_f(cv1);
      sc[rr][u0] = cv0;
      sc[rr][u1] = cv1;
      sh_pk[rr][kk] = pack2bf(hv0, hv1);
    }
    __syncthreads();
  }

  for (int i = tid; i < 16 * FI; i += NW * 32) {
    int rr = i / FI, u = i % FI;
    unsigned pw = sh_pk[rr][u >> 1];
    unsigned short us = (u & 1) ? (unsigned short)(pw >> 16) : (unsigned short)pw;
    hT_out[(size_t)(node0 + rr) * FI + u] = bf2f(us);
  }
}

// ---------------------------------------------------------------------------
// Generic tiled GEMM: out[n][g] = sum_k A[n][k]*W[g][k] + b1[g] + b2[g]
// One wave per 16x16 output tile. Even/odd kt accumulators halve the
// dependent WMMA chain (KT=1 degenerates to a single WMMA + add 0).
// ---------------------------------------------------------------------------
template <int K, int KT>
__global__ __launch_bounds__(32) void gemm_bias_kernel(
    const float* __restrict__ A, const unsigned* __restrict__ wf,
    const float* __restrict__ b1, const float* __restrict__ b2,
    float* __restrict__ out, int Gtot) {
  const int lane = threadIdx.x & 31;
  const int nt = blockIdx.x;
  const int gt = blockIdx.y;
  const int r = lane & 15, half = lane >> 4;
  const float* arow = A + (size_t)(nt * 16 + r) * K;

  v16bf a[KT];
#pragma unroll
  for (int kt = 0; kt < KT; ++kt) a[kt] = a_frag_f32<K>(arow, kt, half);

  v8f acc0 = {};
  v8f acc1 = {};
#pragma unroll
  for (int kt = 0; kt < KT; ++kt) {
    v16bf b = b_frag(wf, gt, kt, KT, lane);
    if (kt & 1)
      acc1 = wmma_bf16(a[kt], b, acc1);
    else
      acc0 = wmma_bf16(a[kt], b, acc0);
  }

  int col = gt * 16 + r;
  float bb = b1[col] + b2[col];
  int mbase = half * 8;
#pragma unroll
  for (int m = 0; m < 8; ++m)
    out[(size_t)(nt * 16 + mbase + m) * Gtot + col] = acc0[m] + acc1[m] + bb;
}

// ---------------------------------------------------------------------------
// Sequential LSTM_o scan over all N "time" steps (batch 1). Single block.
// Per-step matvec h@Whh^T done with WMMA: A is a 16x32 tile whose row 0 is h
// (other rows zero); Whh B-fragments are REGISTER-resident for the whole
// scan (TPW*KT*8 VGPRs/wave). Row 0 of C = acc[0] on lanes 0..15.
// xg already contains x@Wih^T + bih + bhh.
// ---------------------------------------------------------------------------
template <int FO, int NW>
__global__ __launch_bounds__(NW * 32) void lstm_o_scan_wmma(
    const float* __restrict__ xg, const unsigned* __restrict__ fhh,
    float* __restrict__ hs, int Nn) {
  constexpr int G = 4 * FO;
  constexpr int KT = FO / 32;
  constexpr int TPW = (G / 16) / NW;
  __shared__ unsigned sh_pk[FO / 2];  // h packed bf16
  __shared__ float    sc[FO];         // c (f32)
  __shared__ float    sgates[G];

  const int tid = threadIdx.x;
  const int lane = tid & 31;
  const int wave = tid >> 5;
  const int r = lane & 15, half = lane >> 4;

  // weights resident in VGPRs for the entire scan
  v16bf bw[TPW][KT];
#pragma unroll
  for (int j = 0; j < TPW; ++j)
#pragma unroll
    for (int kt = 0; kt < KT; ++kt)
      bw[j][kt] = b_frag(fhh, wave * TPW + j, kt, KT, lane);

  for (int i = tid; i < FO / 2; i += NW * 32) sh_pk[i] = 0u;
  for (int i = tid; i < FO; i += NW * 32) sc[i] = 0.f;
  __syncthreads();

  for (int n = 0; n < Nn; ++n) {
    if (n + 16 < Nn) __builtin_prefetch(xg + (size_t)(n + 16) * G + tid, 0, 0);

    // preload this step's xg slice so global latency overlaps the WMMAs
    float xv[TPW];
#pragma unroll
    for (int j = 0; j < TPW; ++j)
      xv[j] = (lane < 16) ? xg[(size_t)n * G + (wave * TPW + j) * 16 + lane] : 0.f;

    // A fragment: row 0 = h (broadcast ds_load of packed pairs), rows 1..15 = 0
    v16bf a[KT];
#pragma unroll
    for (int kt = 0; kt < KT; ++kt) {
      union { v16bf v; unsigned u[8]; } ua;
#pragma unroll
      for (int d = 0; d < 8; ++d) {
        int e = 2 * d;
        int k0 = kt * 32 + ((e < 8) ? (8 * half + e) : (16 + 8 * half + (e - 8)));
        unsigned val = sh_pk[k0 >> 1];
        ua.u[d] = (r == 0) ? val : 0u;
      }
      a[kt] = ua.v;
    }

    // kt-outer / tile-inner: independent accumulator chains, no hazard NOPs
    v8f acc[TPW] = {};
#pragma unroll
    for (int kt = 0; kt < KT; ++kt)
#pragma unroll
      for (int j = 0; j < TPW; ++j)
        acc[j] = wmma_bf16(a[kt], bw[j][kt], acc[j]);

#pragma unroll
    for (int j = 0; j < TPW; ++j)
      if (lane < 16) sgates[(wave * TPW + j) * 16 + lane] = acc[j][0] + xv[j];
    __syncthreads();

    if (tid < FO) {
      float gi = sgates[tid], gf = sgates[FO + tid];
      float gg = sgates[2 * FO + tid], go = sgates[3 * FO + tid];
      float cv = sigf(gf) * sc[tid] + sigf(gi) * tanh_f(gg);
      float hv = sigf(go) * tanh_f(cv);
      sc[tid] = cv;
      ((unsigned short*)sh_pk)[tid] = f2bf(hv);
      hs[(size_t)n * FO + tid] = hv;
    }
    __syncthreads();
  }
}

// ---------------------------------------------------------------------------
// relu + row L2 normalize (rows of length 128), in place.
// ---------------------------------------------------------------------------
__global__ __launch_bounds__(128) void relu_norm_kernel(float* __restrict__ h) {
  __shared__ float part[4];
  const int tid = threadIdx.x;
  size_t base = (size_t)blockIdx.x * 128;
  float v = fmaxf(h[base + tid], 0.f);
  float s = v * v;
#pragma unroll
  for (int o = 16; o > 0; o >>= 1) s += __shfl_down(s, o, 32);
  if ((tid & 31) == 0) part[tid >> 5] = s;
  __syncthreads();
  float tot = part[0] + part[1] + part[2] + part[3];
  float scale = 1.f / fmaxf(sqrtf(tot), 1e-12f);
  h[base + tid] = v * scale;
}

// ---------------------------------------------------------------------------
extern "C" void kernel_launch(void* const* d_in, const int* in_sizes, int n_in,
                              void* d_out, int out_size, void* d_ws, size_t ws_size,
                              hipStream_t stream) {
  (void)in_sizes; (void)n_in; (void)out_size; (void)ws_size;
  const int N = NNODES;

  const float* x   = (const float*)d_in[0];
  const int*   nbr = (const int*)d_in[1];
  const float* P0[11];
  const float* P1[11];
  for (int i = 0; i < 11; ++i) P0[i] = (const float*)d_in[2 + i];
  for (int i = 0; i < 11; ++i) P1[i] = (const float*)d_in[13 + i];
  // P[.]: 0 Wih_r, 1 Whh_r, 2 bih_r, 3 bhh_r, 4 Wih_o, 5 Whh_o, 6 bih_o,
  //       7 bhh_o, 8 Wlin, 9 blin, 10 bias

  // workspace bump allocator (256B aligned)
  char* base = (char*)d_ws;
  size_t off = 0;
  auto alloc = [&](size_t bytes) -> void* {
    void* p = base + off;
    off = (off + bytes + 255) & ~(size_t)255;
    return p;
  };
  auto fragBytes = [](int G, int KT) -> size_t { return (size_t)G * KT * 64; };

  unsigned* f_ih_r0 = (unsigned*)alloc(fragBytes(64, 1));
  unsigned* f_hh_r0 = (unsigned*)alloc(fragBytes(64, 1));
  unsigned* f_ih_o0 = (unsigned*)alloc(fragBytes(512, 1));
  unsigned* f_hh_o0 = (unsigned*)alloc(fragBytes(512, 4));
  unsigned* f_lin0  = (unsigned*)alloc(fragBytes(128, 4));
  unsigned* f_ih_r1 = (unsigned*)alloc(fragBytes(512, 4));
  unsigned* f_hh_r1 = (unsigned*)alloc(fragBytes(512, 4));
  unsigned* f_ih_o1 = (unsigned*)alloc(fragBytes(256, 4));
  unsigned* f_hh_o1 = (unsigned*)alloc(fragBytes(256, 2));
  unsigned* f_lin1  = (unsigned*)alloc(fragBytes(64, 2));
  float* hagg  = (float*)alloc((size_t)N * 128 * 4);
  float* xg2   = (float*)alloc((size_t)N * 512 * 4);
  float* hsbuf = (float*)alloc((size_t)N * 128 * 4);
  float* h0    = (float*)alloc((size_t)N * 128 * 4);

  // --- pack weights into WMMA B-fragment layout (bf16) ---
  convert_wfrag_kernel<<<dim3(4, 1),  32, 0, stream>>>(P0[0], f_ih_r0, 64, 16, 1);
  convert_wfrag_kernel<<<dim3(4, 1),  32, 0, stream>>>(P0[1], f_hh_r0, 64, 16, 1);
  convert_wfrag_kernel<<<dim3(32, 1), 32, 0, stream>>>(P0[4], f_ih_o0, 512, 16, 1);
  convert_wfrag_kernel<<<dim3(32, 4), 32, 0, stream>>>(P0[5], f_hh_o0, 512, 128, 4);
  convert_wfrag_kernel<<<dim3(8, 4),  32, 0, stream>>>(P0[8], f_lin0, 128, 128, 4);
  convert_wfrag_kernel<<<dim3(32, 4), 32, 0, stream>>>(P1[0], f_ih_r1, 512, 128, 4);
  convert_wfrag_kernel<<<dim3(32, 4), 32, 0, stream>>>(P1[1], f_hh_r1, 512, 128, 4);
  convert_wfrag_kernel<<<dim3(16, 4), 32, 0, stream>>>(P1[4], f_ih_o1, 256, 128, 4);
  convert_wfrag_kernel<<<dim3(16, 2), 32, 0, stream>>>(P1[5], f_hh_o1, 256, 64, 2);
  convert_wfrag_kernel<<<dim3(4, 2),  32, 0, stream>>>(P1[8], f_lin1, 64, 64, 2);

  const int NT = N / 16;  // 2048 node tiles

  // ================= layer 0 (fi=16, fo=128) =================
  lstm_r_kernel<16, 64, 1, 4><<<NT, 4 * 32, 0, stream>>>(
      x, nbr, f_ih_r0, f_hh_r0, P0[2], P0[3], hagg);
  gemm_bias_kernel<16, 1><<<dim3(NT, 32), 32, 0, stream>>>(
      hagg, f_ih_o0, P0[6], P0[7], xg2, 512);
  lstm_o_scan_wmma<128, 8><<<1, 256, 0, stream>>>(xg2, f_hh_o0, hsbuf, N);
  gemm_bias_kernel<128, 4><<<dim3(NT, 8), 32, 0, stream>>>(
      hsbuf, f_lin0, P0[9], P0[10], h0, 128);
  relu_norm_kernel<<<N, 128, 0, stream>>>(h0);

  // ================= layer 1 (fi=128, fo=64) =================
  lstm_r_kernel<128, 512, 4, 8><<<NT, 8 * 32, 0, stream>>>(
      h0, nbr, f_ih_r1, f_hh_r1, P1[2], P1[3], hagg);
  gemm_bias_kernel<128, 4><<<dim3(NT, 16), 32, 0, stream>>>(
      hagg, f_ih_o1, P1[6], P1[7], xg2, 256);
  lstm_o_scan_wmma<64, 8><<<1, 256, 0, stream>>>(xg2, f_hh_o1, hsbuf, N);
  gemm_bias_kernel<64, 2><<<dim3(NT, 4), 32, 0, stream>>>(
      hsbuf, f_lin1, P1[9], P1[10], (float*)d_out, 64);
}
